// DCRBM_84061099917744
// MI455X (gfx1250) — matrix operand order repaired and checked by reading, not compiled, not run on hardware
//
#include <hip/hip_runtime.h>
#include <hip/hip_bf16.h>

typedef __attribute__((ext_vector_type(16))) _Float16 v16h;
typedef __attribute__((ext_vector_type(8)))  _Float16 v8h;
typedef __attribute__((ext_vector_type(4)))  _Float16 v4h;
typedef __attribute__((ext_vector_type(8)))  float    v8f;
typedef __attribute__((ext_vector_type(4)))  int      v4i;

#define BATCH 4096
#define NVIS  1024
#define NHID  4096
#define NCOND 512

// LDS strides in halves: 80-byte rows -> every 16B chunk aligned, and the 16
// row-readers per lane-half hit 16 distinct banks.
#define A_STRIDE  40   // A tile: 128 rows x 32 K
#define BT_STRIDE 40   // B tile transposed: 64 n-rows x 32 K

// ---------------------------------------------------------------------------
// Async global->LDS (GLOBAL_LOAD_ASYNC_TO_LDS_B128, tracked on ASYNCcnt).
// Signature (from hipcc diagnostic): (v4i AS1*, v4i AS3*, imm offset, imm cpol)
// ---------------------------------------------------------------------------
#if __has_builtin(__builtin_amdgcn_global_load_async_to_lds_b128)
#define HAVE_ASYNC_LDS 1
typedef __attribute__((address_space(1))) v4i gv4i;
typedef __attribute__((address_space(3))) v4i lv4i;
__device__ inline void async_ld_b128(const void* g, void* l) {
    __builtin_amdgcn_global_load_async_to_lds_b128((gv4i*)g, (lv4i*)l, 0, 0);
}
#else
#define HAVE_ASYNC_LDS 0
#endif

__device__ inline void wait_async_zero() {
#if HAVE_ASYNC_LDS
    asm volatile("s_wait_asynccnt 0x0" ::: "memory");
#endif
}

// ---------------------------------------------------------------------------
// f32 -> f16 conversion
// ---------------------------------------------------------------------------
__global__ __launch_bounds__(256) void f32_to_f16_kernel(const float* __restrict__ in,
                                                         _Float16* __restrict__ out, int n) {
    int i = blockIdx.x * 256 + threadIdx.x;
    if (i < n) out[i] = (_Float16)in[i];
}

// ---------------------------------------------------------------------------
// Wide fragment loads (ds_load_b128) per ISA 16x16x32-f16 layouts
// ---------------------------------------------------------------------------
// A: lane l -> row m=l&15; halves = K [8*hi, 8*hi+8) ++ K [16+8*hi, 24+8*hi)
__device__ inline v16h frag_a(const _Float16* At, int lane) {
    const int m  = lane & 15;
    const int hi = lane >> 4;
    const _Float16* row = At + m * A_STRIDE;
    const v8h lo = *(const v8h*)(row + 8 * hi);
    const v8h hb = *(const v8h*)(row + 16 + 8 * hi);
    return __builtin_shufflevector(lo, hb, 0, 1, 2, 3, 4, 5, 6, 7,
                                           8, 9, 10, 11, 12, 13, 14, 15);
}

// B (from transposed LDS tile BtT[n][k]): lane l -> col n=l&15;
// halves = K [8*hi, 8*hi+8) ++ K [16+8*hi, 24+8*hi)
__device__ inline v16h frag_b(const _Float16* BtT, int lane, int ncol0) {
    const int n  = lane & 15;
    const int hi = lane >> 4;
    const _Float16* row = BtT + (ncol0 + n) * BT_STRIDE + 8 * hi;
    const v8h lo = *(const v8h*)(row);
    const v8h hb = *(const v8h*)(row + 16);
    return __builtin_shufflevector(lo, hb, 0, 1, 2, 3, 4, 5, 6, 7,
                                           8, 9, 10, 11, 12, 13, 14, 15);
}

// 8 WMMAs per wave per K-step: 32x64 wave tile (2 M-subtiles x 4 N-subtiles)
__device__ inline void compute_step(const _Float16* At, const _Float16* BtT,
                                    v8f* acc, int lane, int w) {
    const v16h a0 = frag_a(At + (w * 32) * A_STRIDE, lane);
    const v16h a1 = frag_a(At + (w * 32 + 16) * A_STRIDE, lane);
#pragma unroll
    for (int nt = 0; nt < 4; ++nt) {
        const v16h b = frag_b(BtT, lane, nt * 16);
        acc[nt] = __builtin_amdgcn_wmma_f32_16x16x32_f16(
            false, a0, false, b, (short)0, acc[nt], false, false);
        acc[4 + nt] = __builtin_amdgcn_wmma_f32_16x16x32_f16(
            false, a1, false, b, (short)0, acc[4 + nt], false, false);
    }
}

// A tile fill: 128x32 halves = 512 b128 transfers, 4 per thread.
__device__ inline void fill_a_tile(_Float16* At, const _Float16* __restrict__ A,
                                   int lda, int m0, int k0, int t) {
#pragma unroll
    for (int i = 0; i < 4; ++i) {
        const int d = t + i * 128;
        const int row = d >> 2, c4 = d & 3;
#if HAVE_ASYNC_LDS
        async_ld_b128(A + (size_t)(m0 + row) * lda + k0 + 8 * c4,
                      At + row * A_STRIDE + 8 * c4);
#else
        ((uint4*)(At + row * A_STRIDE))[c4] =
            ((const uint4*)(A + (size_t)(m0 + row) * lda + k0))[c4];
#endif
    }
}

// ---------------------------------------------------------------------------
// K-step: B source is row-major [K x N] -> transpose-scatter into BtT
// ---------------------------------------------------------------------------
__device__ inline void kstep_rowmajorB(_Float16* At, _Float16* BtT,
                                       const _Float16* __restrict__ A, int lda,
                                       const _Float16* __restrict__ B, int ldb,
                                       int m0, int n0, int k0, int kend,
                                       v8f* acc, int t, int lane, int w) {
    __syncthreads();
    fill_a_tile(At, A, lda, m0, k0, t);
    // B tile: 32x64 halves, read v4h along n, scatter-transpose into BtT[n][k]
#pragma unroll
    for (int i = 0; i < 4; ++i) {
        const int d = t + i * 128;
        const int k = d >> 4, q = d & 15;
        const v4h x = *(const v4h*)(B + (size_t)(k0 + k) * ldb + n0 + 4 * q);
#pragma unroll
        for (int j = 0; j < 4; ++j) BtT[(4 * q + j) * BT_STRIDE + k] = x[j];
    }
    if (k0 + 32 < kend) {  // prefetch next K tile (global_prefetch_b8)
        __builtin_prefetch(A + (size_t)(m0 + t) * lda + (k0 + 32), 0, 1);
        if (t < 32) __builtin_prefetch(B + (size_t)(k0 + 32 + t) * ldb + n0, 0, 1);
    }
    wait_async_zero();
    __syncthreads();
    compute_step(At, BtT, acc, lane, w);
}

// K-step: B operand is W^T, i.e. BtT[n][k] = W[n0+n][k0+k] -> contiguous fill
__device__ inline void kstep_transB(_Float16* At, _Float16* BtT,
                                    const _Float16* __restrict__ A, int lda,
                                    const _Float16* __restrict__ Wsrc, int ldw,
                                    int m0, int n0, int k0, int kend,
                                    v8f* acc, int t, int lane, int w) {
    __syncthreads();
    fill_a_tile(At, A, lda, m0, k0, t);
    // BtT rows are contiguous runs of W's rows: 64x32 halves = 256 b128
#pragma unroll
    for (int i = 0; i < 2; ++i) {
        const int d = t + i * 128;
        const int n = d >> 2, c4 = d & 3;
#if HAVE_ASYNC_LDS
        async_ld_b128(Wsrc + (size_t)(n0 + n) * ldw + k0 + 8 * c4,
                      BtT + n * BT_STRIDE + 8 * c4);
#else
        ((uint4*)(BtT + n * BT_STRIDE))[c4] =
            ((const uint4*)(Wsrc + (size_t)(n0 + n) * ldw + k0))[c4];
#endif
    }
    if (k0 + 32 < kend) {
        __builtin_prefetch(A + (size_t)(m0 + t) * lda + (k0 + 32), 0, 1);
        if (t < 64) __builtin_prefetch(Wsrc + (size_t)(n0 + t) * ldw + (k0 + 32), 0, 1);
    }
    wait_async_zero();
    __syncthreads();
    compute_step(At, BtT, acc, lane, w);
}

// ---------------------------------------------------------------------------
// GEMM 1: Hh = f16( noise < sigmoid(v@W + u@B + c) )   [BATCH x NHID]
// block tile 128x64, 4 waves, wave tile 32x64
// ---------------------------------------------------------------------------
__global__ __launch_bounds__(128) void gemm_h_kernel(
    const _Float16* __restrict__ Vh,     // [BATCH x NVIS]
    const _Float16* __restrict__ Uh,     // [BATCH x NCOND]
    const _Float16* __restrict__ Wh,     // [NVIS x NHID]
    const _Float16* __restrict__ Bch,    // [NCOND x NHID]
    const float* __restrict__ cbias,     // [NHID]
    const float* __restrict__ noise,     // [BATCH x NHID]
    _Float16* __restrict__ Hh)           // [BATCH x NHID]
{
    __shared__ __align__(16) _Float16 At[128 * A_STRIDE];
    __shared__ __align__(16) _Float16 BtT[64 * BT_STRIDE];
    const int m0 = blockIdx.y * 128;
    const int n0 = blockIdx.x * 64;
    const int t = threadIdx.x;
    const int lane = t & 31;
    const int w = t >> 5;

    v8f acc[8];
#pragma unroll
    for (int i = 0; i < 8; ++i) acc[i] = (v8f){0, 0, 0, 0, 0, 0, 0, 0};

    for (int k0 = 0; k0 < NVIS; k0 += 32)
        kstep_rowmajorB(At, BtT, Vh, NVIS, Wh, NHID, m0, n0, k0, NVIS, acc, t, lane, w);
    for (int k0 = 0; k0 < NCOND; k0 += 32)
        kstep_rowmajorB(At, BtT, Uh, NCOND, Bch, NHID, m0, n0, k0, NCOND, acc, t, lane, w);

    const int hi = lane >> 4, nl = lane & 15;
#pragma unroll
    for (int mt = 0; mt < 2; ++mt) {
#pragma unroll
        for (int nt = 0; nt < 4; ++nt) {
            const int col = n0 + nt * 16 + nl;
            const float cb = cbias[col];
#pragma unroll
            for (int r = 0; r < 8; ++r) {
                const int row = m0 + w * 32 + mt * 16 + r + 8 * hi;
                const float pre = acc[mt * 4 + nt][r] + cb;
                const float s = 1.0f / (1.0f + __expf(-pre));
                const float h = (noise[(size_t)row * NHID + col] < s) ? 1.0f : 0.0f;
                Hh[(size_t)row * NHID + col] = (_Float16)h;
            }
        }
    }
}

// ---------------------------------------------------------------------------
// GEMM 2: pre = h1_sample @ W.T + u @ A + b   [BATCH x NVIS], f32
// ---------------------------------------------------------------------------
__global__ __launch_bounds__(128) void gemm_v_kernel(
    const _Float16* __restrict__ Hh,     // [BATCH x NHID]
    const _Float16* __restrict__ Uh,     // [BATCH x NCOND]
    const _Float16* __restrict__ Wh,     // [NVIS x NHID] (used transposed)
    const _Float16* __restrict__ Ah,     // [NCOND x NVIS]
    const float* __restrict__ bbias,     // [NVIS]
    float* __restrict__ pre)             // [BATCH x NVIS]
{
    __shared__ __align__(16) _Float16 At[128 * A_STRIDE];
    __shared__ __align__(16) _Float16 BtT[64 * BT_STRIDE];
    const int m0 = blockIdx.y * 128;
    const int n0 = blockIdx.x * 64;
    const int t = threadIdx.x;
    const int lane = t & 31;
    const int w = t >> 5;

    v8f acc[8];
#pragma unroll
    for (int i = 0; i < 8; ++i) acc[i] = (v8f){0, 0, 0, 0, 0, 0, 0, 0};

    for (int k0 = 0; k0 < NHID; k0 += 32)
        kstep_transB(At, BtT, Hh, NHID, Wh, NHID, m0, n0, k0, NHID, acc, t, lane, w);
    for (int k0 = 0; k0 < NCOND; k0 += 32)
        kstep_rowmajorB(At, BtT, Uh, NCOND, Ah, NVIS, m0, n0, k0, NCOND, acc, t, lane, w);

    const int hi = lane >> 4, nl = lane & 15;
#pragma unroll
    for (int mt = 0; mt < 2; ++mt) {
#pragma unroll
        for (int nt = 0; nt < 4; ++nt) {
            const int col = n0 + nt * 16 + nl;
            const float bb = bbias[col];
#pragma unroll
            for (int r = 0; r < 8; ++r) {
                const int row = m0 + w * 32 + mt * 16 + r + 8 * hi;
                pre[(size_t)row * NVIS + col] = acc[mt * 4 + nt][r] + bb;
            }
        }
    }
}

// ---------------------------------------------------------------------------
// Row argmax (first-index tie rule) -> one-hot (softmax is monotone, skip it)
// ---------------------------------------------------------------------------
__global__ __launch_bounds__(256) void argmax_onehot_kernel(
    const float* __restrict__ pre, float* __restrict__ out) {
    const int row = blockIdx.x;
    const int t = threadIdx.x;
    const float* p = pre + (size_t)row * NVIS;

    float best = -3.402823466e+38f;
    int bidx = NVIS;
    for (int n = t; n < NVIS; n += 256) {
        const float v = p[n];
        if (v > best || (v == best && n < bidx)) { best = v; bidx = n; }
    }
    __shared__ float sv[256];
    __shared__ int si[256];
    sv[t] = best; si[t] = bidx;
    __syncthreads();
    for (int s = 128; s > 0; s >>= 1) {
        if (t < s) {
            if (sv[t + s] > sv[t] || (sv[t + s] == sv[t] && si[t + s] < si[t])) {
                sv[t] = sv[t + s]; si[t] = si[t + s];
            }
        }
        __syncthreads();
    }
    const int idx = si[0];
    float* o = out + (size_t)row * NVIS;
    for (int n = t; n < NVIS; n += 256) o[n] = (n == idx) ? 1.0f : 0.0f;
}

// ---------------------------------------------------------------------------
// Launch
// ---------------------------------------------------------------------------
extern "C" void kernel_launch(void* const* d_in, const int* in_sizes, int n_in,
                              void* d_out, int out_size, void* d_ws, size_t ws_size,
                              hipStream_t stream) {
    const float* visible = (const float*)d_in[0];  // [4096 x 1024]
    const float* cond    = (const float*)d_in[1];  // [4096 x 512]
    const float* noise   = (const float*)d_in[2];  // [4096 x 4096]
    const float* W       = (const float*)d_in[3];  // [1024 x 4096]
    const float* bbias   = (const float*)d_in[4];  // [1 x 1024]
    const float* cbias   = (const float*)d_in[5];  // [1 x 4096]
    const float* A       = (const float*)d_in[6];  // [512 x 1024]
    const float* B       = (const float*)d_in[7];  // [512 x 4096]
    float* out = (float*)d_out;                    // [4096 x 1024] one-hot

    // Workspace carve (bytes)
    char* ws = (char*)d_ws;
    _Float16* Vh  = (_Float16*)(ws + 0);                 // 8  MiB
    _Float16* Uh  = (_Float16*)(ws + 8388608);           // 4  MiB
    _Float16* Wh  = (_Float16*)(ws + 12582912);          // 8  MiB
    _Float16* Ah  = (_Float16*)(ws + 20971520);          // 1  MiB
    _Float16* Bch = (_Float16*)(ws + 22020096);          // 4  MiB
    _Float16* Hh  = (_Float16*)(ws + 26214400);          // 32 MiB
    float*    pre = (float*)   (ws + 59768832);          // 16 MiB

    auto cvt = [&](const float* src, _Float16* dst, int n) {
        f32_to_f16_kernel<<<(n + 255) / 256, 256, 0, stream>>>(src, dst, n);
    };
    cvt(visible, Vh, BATCH * NVIS);
    cvt(cond,    Uh, BATCH * NCOND);
    cvt(W,       Wh, NVIS * NHID);
    cvt(A,       Ah, NCOND * NVIS);
    cvt(B,       Bch, NCOND * NHID);

    // GEMM1: [4096 x 4096], block tile 128x64
    gemm_h_kernel<<<dim3(NHID / 64, BATCH / 128), 128, 0, stream>>>(
        Vh, Uh, Wh, Bch, cbias, noise, Hh);

    // GEMM2: [4096 x 1024], block tile 128x64
    gemm_v_kernel<<<dim3(NVIS / 64, BATCH / 128), 128, 0, stream>>>(
        Hh, Uh, Wh, Ah, bbias, pre);

    // Argmax -> one-hot
    argmax_onehot_kernel<<<BATCH, 256, 0, stream>>>(pre, out);
}
